// ReNet_layer_37812892074625
// MI455X (gfx1250) — matrix operand order repaired and checked by reading, not compiled, not run on hardware
//
#include <hip/hip_runtime.h>
#include <math.h>

// ---------------------------------------------------------------------------
// ReNet layer for MI455X (gfx1250, wave32, WMMA).
//
//   1) renet_vgru : vertical bidirectional GRU (hidden 64/dir) over m-axis.
//      grid (8 n-blocks, 8 batch, 2 dirs), 128 threads (4 waves).
//      All Wih/Whh B-fragments held in VGPRs, hidden state ping-pongs through
//      LDS, per-step GEMMs via v_wmma_f32_16x16x32_f16 with separate
//      x-side / h-side accumulators (GRU applies r *inside* the n-gate h-term).
//   2) renet_hgru : horizontal bidirectional GRU (hidden 128/dir), 8 waves;
//      input tile staged global->LDS with async b128 copies (ASYNCcnt).
//   3) renet_transpose : [n,m,b,256] f32 -> [b,256,m,n] f32 final layout.
//
// Gate nonlinearities use the CDNA5 hardware TANH transcendental when the
// builtin is exposed (sigmoid(x) = 0.5 + 0.5*tanh(0.5x)).
// ---------------------------------------------------------------------------

typedef __attribute__((ext_vector_type(16))) _Float16 v16h;
typedef __attribute__((ext_vector_type(8)))  _Float16 v8h;
typedef __attribute__((ext_vector_type(8)))  float    v8f;
typedef __attribute__((ext_vector_type(4)))  int      v4i;

#if __has_builtin(__builtin_amdgcn_tanhf)
#define FAST_TANH(x) __builtin_amdgcn_tanhf(x)
#define HAVE_HW_TANH 1
#elif __has_builtin(__builtin_amdgcn_tanh_f32)
#define FAST_TANH(x) __builtin_amdgcn_tanh_f32(x)
#define HAVE_HW_TANH 1
#else
#define FAST_TANH(x) tanhf(x)
#define HAVE_HW_TANH 0
#endif

__device__ __forceinline__ float fast_sigmoid(float x) {
#if HAVE_HW_TANH
  return 0.5f + 0.5f * FAST_TANH(0.5f * x);   // 1 trans + 2 VALU
#else
  return __builtin_amdgcn_rcpf(1.0f + __expf(-x));  // v_rcp, not IEEE div
#endif
}

// Async global->LDS staging (16B per lane), tracked by ASYNCcnt.
// Probe-confirmed signature: (int4 AS(1)*, int4 AS(3)*, imm offset, imm cpol).
#if __has_builtin(__builtin_amdgcn_global_load_async_to_lds_b128)
#define HAVE_ASYNC_LDS 1
typedef __attribute__((address_space(1))) v4i glb_v4i;
typedef __attribute__((address_space(3))) v4i lds_v4i;
#else
#define HAVE_ASYNC_LDS 0
#endif

__device__ __forceinline__ void wait_asynccnt0() {
#if __has_builtin(__builtin_amdgcn_s_wait_asynccnt)
  __builtin_amdgcn_s_wait_asynccnt(0);
#else
  asm volatile("s_wait_asynccnt 0x0" ::: "memory");
#endif
}

// A-fragment (16x32 f16, M=seq rows, K) from an LDS tile with row stride ldh
// (halves, multiple of 8 so the 16B vector loads stay aligned).
// ISA layout: lanes 0-15 hold K = {kofs+0..7, kofs+16..23},
//             lanes 16-31 hold K = {kofs+8..15, kofs+24..31}; M = lane%16.
__device__ __forceinline__ v16h load_a_lds(const _Float16* base, int ldh,
                                           int row, int kofs, int lane) {
  const int k0 = kofs + ((lane & 16) ? 8 : 0);
  const _Float16* p = base + row * ldh + k0;
  v8h lo = *(const v8h*)(p);
  v8h hi = *(const v8h*)(p + 16);
  v16h a;
#pragma unroll
  for (int i = 0; i < 8; ++i) { a[i] = lo[i]; a[i + 8] = hi[i]; }
  return a;
}

// B-fragment (32x16 f16, K x N) for out = act @ W^T:  B[k][n] = W[nrow][k].
// lane%16 = N column; kbase = kofs + 16*half; elements follow A's order:
// K = {kofs+8*half+0..7, kofs+16+8*half+0..7}.
__device__ __forceinline__ v16h load_b_gmem(const float* __restrict__ W,
                                            int ldw, int nrow, int kbase) {
  const int hh = (kbase & 16) >> 1;
  const float* p = W + (size_t)nrow * ldw + (kbase - (kbase & 16)) + hh;
  v16h b;
#pragma unroll
  for (int i = 0; i < 8; ++i) { b[i] = (_Float16)p[i]; b[i + 8] = (_Float16)p[i + 16]; }
  return b;
}

// ---------------------------------------------------------------------------
// Vertical pass: sequences = (n, b), time = m, input C=128, hidden H=64/dir.
// Writes f16 v-buffer laid out [m][n][b][128ch] (fwd -> ch 0..63, bwd 64..127).
// ---------------------------------------------------------------------------
__global__ __launch_bounds__(128) void renet_vgru(
    const float* __restrict__ x,
    const float* __restrict__ wih_f, const float* __restrict__ whh_f,
    const float* __restrict__ bih_f, const float* __restrict__ bhh_f,
    const float* __restrict__ wih_b, const float* __restrict__ whh_b,
    const float* __restrict__ bih_b, const float* __restrict__ bhh_b,
    _Float16* __restrict__ vbuf) {
  constexpr int C = 128, Mlen = 128, Nn = 128, Bb = 8, H = 64;
  constexpr int LDX = C + 16;  // 144 halves (288B rows, 16B aligned)
  constexpr int LDH = H + 16;  // 80 halves  (160B rows, 16B aligned)
  __shared__ alignas(16) _Float16 Xs[16 * LDX];
  __shared__ alignas(16) _Float16 Hs[16 * LDH];

  const int tid  = threadIdx.x;
  const int lane = tid & 31;
  const int wv   = tid >> 5;                 // wave 0..3 owns hidden cols 16w..16w+15
  const int l15  = lane & 15;
  const int kh   = (lane & 16) ? 16 : 0;     // B-fragment half-wave K offset

  const int n0  = blockIdx.x * 16;
  const int b   = blockIdx.y;
  const int dir = blockIdx.z;

  const float* Wih = dir ? wih_b : wih_f;
  const float* Whh = dir ? whh_b : whh_f;
  const float* Bih = dir ? bih_b : bih_f;
  const float* Bhh = dir ? bhh_b : bhh_f;

  for (int i = tid; i < 16 * LDH; i += 128) Hs[i] = (_Float16)0.0f;

  // Gate tiles over N=192: {r:0..3, z:4..7, n:8..11}; wave w -> {w, w+4, w+8}
  // so gate math stays wave-local on matching hidden columns.
  v16h bx[3][4];  // x-side: K = C = 128 -> 4 k-iters
  v16h bh[3][2];  // h-side: K = H = 64 -> 2 k-iters
#pragma unroll
  for (int g = 0; g < 3; ++g) {
    const int nrow = (g * 4 + wv) * 16 + l15;
#pragma unroll
    for (int ki = 0; ki < 4; ++ki) bx[g][ki] = load_b_gmem(Wih, C, nrow, 32 * ki + kh);
#pragma unroll
    for (int ki = 0; ki < 2; ++ki) bh[g][ki] = load_b_gmem(Whh, H, nrow, 32 * ki + kh);
  }
  const int j = wv * 16 + l15;  // this lane's hidden column
  const float b_ir = Bih[j], b_iz = Bih[H + j], b_in = Bih[2 * H + j];
  const float b_hr = Bhh[j], b_hz = Bhh[H + j], b_hn = Bhh[2 * H + j];
  float hreg[8] = {0.f, 0.f, 0.f, 0.f, 0.f, 0.f, 0.f, 0.f};
  __syncthreads();

  for (int t = 0; t < Mlen; ++t) {
    const int tt = dir ? (Mlen - 1 - t) : t;
    // Stage x_t tile: Xs[row][ch] = x[b][ch][tt][n0+row] (coalesced along n).
#pragma unroll 4
    for (int idx = tid; idx < 16 * C; idx += 128) {
      const int row = idx & 15, ch = idx >> 4;
      Xs[row * LDX + ch] =
          (_Float16)x[(((size_t)b * C + ch) * Mlen + tt) * Nn + (n0 + row)];
    }
    __syncthreads();

    v8f gx[3] = {}, gh[3] = {};
#pragma unroll
    for (int ki = 0; ki < 4; ++ki) {
      const v16h a = load_a_lds(Xs, LDX, l15, 32 * ki, lane);
#pragma unroll
      for (int g = 0; g < 3; ++g)
        gx[g] = __builtin_amdgcn_wmma_f32_16x16x32_f16(false, a, false, bx[g][ki],
                                                       (short)0, gx[g], false, false);
    }
#pragma unroll
    for (int ki = 0; ki < 2; ++ki) {
      const v16h a = load_a_lds(Hs, LDH, l15, 32 * ki, lane);
#pragma unroll
      for (int g = 0; g < 3; ++g)
        gh[g] = __builtin_amdgcn_wmma_f32_16x16x32_f16(false, a, false, bh[g][ki],
                                                       (short)0, gh[g], false, false);
    }

    // GRU gate math (r, z, n order as in PyTorch).
#pragma unroll
    for (int r = 0; r < 8; ++r) {
      const float rr = fast_sigmoid(gx[0][r] + b_ir + gh[0][r] + b_hr);
      const float zz = fast_sigmoid(gx[1][r] + b_iz + gh[1][r] + b_hz);
      const float nn = FAST_TANH(gx[2][r] + b_in + rr * (gh[2][r] + b_hn));
      hreg[r] = (1.0f - zz) * nn + zz * hreg[r];
    }
    __syncthreads();  // all waves done reading Hs/Xs
#pragma unroll
    for (int r = 0; r < 8; ++r) {
      const int mrow = r + ((lane >> 4) << 3);
      const _Float16 hv = (_Float16)hreg[r];
      Hs[mrow * LDH + j] = hv;
      vbuf[(((size_t)tt * Nn + (n0 + mrow)) * Bb + b) * C + dir * H + j] = hv;
    }
    __syncthreads();
  }
}

// ---------------------------------------------------------------------------
// Horizontal pass: sequences = (m, b), time = n, input C=128 (v), hidden 128.
// Writes f32 hout laid out [n][m][b][256] (coalesced); transposed afterwards.
// ---------------------------------------------------------------------------
__global__ __launch_bounds__(256) void renet_hgru(
    const _Float16* __restrict__ vbuf,
    const float* __restrict__ wih_f, const float* __restrict__ whh_f,
    const float* __restrict__ bih_f, const float* __restrict__ bhh_f,
    const float* __restrict__ wih_b, const float* __restrict__ whh_b,
    const float* __restrict__ bih_b, const float* __restrict__ bhh_b,
    float* __restrict__ hout) {
  constexpr int C = 128, Mlen = 128, Nn = 128, Bb = 8, H = 128;
  constexpr int LDX = C + 16;  // 144
  constexpr int LDH = H + 16;  // 144
  __shared__ alignas(16) _Float16 Xs[16 * LDX];
  __shared__ alignas(16) _Float16 Hs[16 * LDH];

  const int tid  = threadIdx.x;
  const int lane = tid & 31;
  const int wv   = tid >> 5;                 // 8 waves: cols 16w..16w+15 of H=128
  const int l15  = lane & 15;
  const int kh   = (lane & 16) ? 16 : 0;

  const int m0  = blockIdx.x * 16;
  const int b   = blockIdx.y;
  const int dir = blockIdx.z;

  const float* Wih = dir ? wih_b : wih_f;
  const float* Whh = dir ? whh_b : whh_f;
  const float* Bih = dir ? bih_b : bih_f;
  const float* Bhh = dir ? bhh_b : bhh_f;

  for (int i = tid; i < 16 * LDH; i += 256) Hs[i] = (_Float16)0.0f;

  // Gate tiles over N=384: {r:0..7, z:8..15, n:16..23}; wave w -> {w, w+8, w+16}.
  v16h bx[3][4], bh[3][4];
#pragma unroll
  for (int g = 0; g < 3; ++g) {
    const int nrow = (g * 8 + wv) * 16 + l15;
#pragma unroll
    for (int ki = 0; ki < 4; ++ki) {
      bx[g][ki] = load_b_gmem(Wih, C, nrow, 32 * ki + kh);
      bh[g][ki] = load_b_gmem(Whh, H, nrow, 32 * ki + kh);
    }
  }
  const int j = wv * 16 + l15;
  const float b_ir = Bih[j], b_iz = Bih[H + j], b_in = Bih[2 * H + j];
  const float b_hr = Bhh[j], b_hz = Bhh[H + j], b_hn = Bhh[2 * H + j];
  float hreg[8] = {0.f, 0.f, 0.f, 0.f, 0.f, 0.f, 0.f, 0.f};

  // Per-thread staging addresses: one 16B chunk per thread covers the 4KB tile.
  const int srow = tid >> 4;          // 0..15
  const int sch0 = (tid & 15) * 8;    // half index, 16B chunks
  __syncthreads();

  for (int t = 0; t < Nn; ++t) {
    const int tt = dir ? (Nn - 1 - t) : t;
    // Stage v_t tile: Xs[row][ch] = vbuf[m0+row][tt][b][ch] (rows are 256B).
    {
      const _Float16* gp =
          vbuf + (((size_t)(m0 + srow) * Nn + tt) * Bb + b) * C + sch0;
      _Float16* lp = &Xs[srow * LDX + sch0];
#if HAVE_ASYNC_LDS
      __builtin_amdgcn_global_load_async_to_lds_b128(
          (glb_v4i*)(void*)gp, (lds_v4i*)(void*)lp, 0, 0);
      wait_asynccnt0();
#else
      *(v8h*)lp = *(const v8h*)gp;
#endif
    }
    __syncthreads();

    v8f gx[3] = {}, gh[3] = {};
#pragma unroll
    for (int ki = 0; ki < 4; ++ki) {
      const v16h ax = load_a_lds(Xs, LDX, l15, 32 * ki, lane);
      const v16h ah = load_a_lds(Hs, LDH, l15, 32 * ki, lane);
#pragma unroll
      for (int g = 0; g < 3; ++g) {
        gx[g] = __builtin_amdgcn_wmma_f32_16x16x32_f16(false, ax, false, bx[g][ki],
                                                       (short)0, gx[g], false, false);
        gh[g] = __builtin_amdgcn_wmma_f32_16x16x32_f16(false, ah, false, bh[g][ki],
                                                       (short)0, gh[g], false, false);
      }
    }

#pragma unroll
    for (int r = 0; r < 8; ++r) {
      const float rr = fast_sigmoid(gx[0][r] + b_ir + gh[0][r] + b_hr);
      const float zz = fast_sigmoid(gx[1][r] + b_iz + gh[1][r] + b_hz);
      const float nn = FAST_TANH(gx[2][r] + b_in + rr * (gh[2][r] + b_hn));
      hreg[r] = (1.0f - zz) * nn + zz * hreg[r];
    }
    __syncthreads();
#pragma unroll
    for (int r = 0; r < 8; ++r) {
      const int mrow = r + ((lane >> 4) << 3);
      Hs[mrow * LDH + j] = (_Float16)hreg[r];
      hout[(((size_t)tt * Mlen + (m0 + mrow)) * Bb + b) * 256 + dir * H + j] = hreg[r];
    }
    __syncthreads();
  }
}

// [n][m][b][256] -> [b][256][m][n]
__global__ __launch_bounds__(256) void renet_transpose(
    const float* __restrict__ hout, float* __restrict__ out) {
  const size_t idx = (size_t)blockIdx.x * 256 + threadIdx.x;  // 2^25 elements
  const int n  = (int)(idx & 127);
  const int m  = (int)((idx >> 7) & 127);
  const int ch = (int)((idx >> 14) & 255);
  const int b  = (int)(idx >> 22);
  out[idx] = hout[(((size_t)n * 128 + m) * 8 + b) * 256 + ch];
}

extern "C" void kernel_launch(void* const* d_in, const int* in_sizes, int n_in,
                              void* d_out, int out_size, void* d_ws, size_t ws_size,
                              hipStream_t stream) {
  (void)in_sizes; (void)n_in; (void)out_size; (void)ws_size;
  const float* x      = (const float*)d_in[0];
  const float* v_wihf = (const float*)d_in[1];
  const float* v_whhf = (const float*)d_in[2];
  const float* v_bihf = (const float*)d_in[3];
  const float* v_bhhf = (const float*)d_in[4];
  const float* v_wihb = (const float*)d_in[5];
  const float* v_whhb = (const float*)d_in[6];
  const float* v_bihb = (const float*)d_in[7];
  const float* v_bhhb = (const float*)d_in[8];
  const float* h_wihf = (const float*)d_in[9];
  const float* h_whhf = (const float*)d_in[10];
  const float* h_bihf = (const float*)d_in[11];
  const float* h_bhhf = (const float*)d_in[12];
  const float* h_wihb = (const float*)d_in[13];
  const float* h_whhb = (const float*)d_in[14];
  const float* h_bihb = (const float*)d_in[15];
  const float* h_bhhb = (const float*)d_in[16];

  // Workspace: v (f16, 128*128*8*128 halves = 32 MiB) then
  //            hout (f32, 128*128*8*256 floats = 128 MiB).
  _Float16* vbuf = (_Float16*)d_ws;
  float* hbuf = (float*)((char*)d_ws + (size_t)32 * 1024 * 1024);

  renet_vgru<<<dim3(8, 8, 2), 128, 0, stream>>>(
      x, v_wihf, v_whhf, v_bihf, v_bhhf, v_wihb, v_whhb, v_bihb, v_bhhb, vbuf);
  renet_hgru<<<dim3(8, 8, 2), 256, 0, stream>>>(
      vbuf, h_wihf, h_whhf, h_bihf, h_bhhf, h_wihb, h_whhb, h_bihb, h_bhhb, hbuf);
  renet_transpose<<<(33554432 / 256), 256, 0, stream>>>(hbuf, (float*)d_out);
}
